// LocalEmbeddingLayer_60954175864839
// MI455X (gfx1250) — matrix operand order repaired and checked by reading, not compiled, not run on hardware
//
#include <hip/hip_runtime.h>
#include <math.h>

// MI455X / gfx1250 implementation.
// Pipeline: knn<3> -> edge-MLP block0 (f16 WMMA) -> knn<64> -> edge-MLP block1 -> mask.
// GEMMs use v_wmma_f32_16x16x32_f16 (f32 accumulate). KNN is VALU with LDS-tiled
// candidates and a fully unrolled register top-17 insertion sort.

typedef __attribute__((ext_vector_type(16))) _Float16 v16h;
typedef __attribute__((ext_vector_type(8)))  _Float16 v8h;
typedef __attribute__((ext_vector_type(8)))  float    v8f;

#define BATCH 64
#define NTOT  512
#define KNN   16

__device__ __forceinline__ float gelu_exact(float x) {
    return 0.5f * x * (1.0f + erff(x * 0.7071067811865476f));
}

// ---------------------------------------------------------------------------
// KNN: for each (b,n) find 17 smallest squared distances (incl. self), emit
// indices 1..16. Coord shift 999*(mask==0) folded in on load.
// ---------------------------------------------------------------------------
template<int F>
__global__ __launch_bounds__(256) void knn_kernel(const float* __restrict__ src,
                                                  const int*   __restrict__ mask,
                                                  int*         __restrict__ idxOut) {
    constexpr int TILE = 128;
    __shared__ float sP[TILE * F];

    const int b = blockIdx.x / (NTOT / 256);
    const int n = (blockIdx.x % (NTOT / 256)) * 256 + threadIdx.x;
    const float* srcB  = src  + (size_t)b * NTOT * F;
    const int*   maskB = mask + b * NTOT;

    float myp[F];
    {
        const float sh = (maskB[n] == 0) ? 999.0f : 0.0f;
        #pragma unroll
        for (int f = 0; f < F; ++f) myp[f] = srcB[n * F + f] + sh;
    }

    float bd[KNN + 1];
    int   bi[KNN + 1];
    #pragma unroll
    for (int j = 0; j <= KNN; ++j) { bd[j] = 3.4e38f; bi[j] = 0; }

    for (int t0 = 0; t0 < NTOT; t0 += TILE) {
        __syncthreads();
        for (int e = threadIdx.x; e < TILE * F; e += 256) {
            const int m = t0 + e / F;
            const int f = e % F;
            const float sh = (maskB[m] == 0) ? 999.0f : 0.0f;
            sP[e] = srcB[m * F + f] + sh;
        }
        __syncthreads();
        for (int t = 0; t < TILE; ++t) {
            float d = 1e-5f;
            #pragma unroll
            for (int f = 0; f < F; ++f) {
                const float df = myp[f] - sP[t * F + f];
                d += df * df;
            }
            if (d < bd[KNN]) {   // unrolled ascending insertion (stable: ties keep lower idx)
                float cd = d; int ci = t0 + t;
                #pragma unroll
                for (int j = 0; j <= KNN; ++j) {
                    if (cd < bd[j]) {
                        float td = bd[j]; bd[j] = cd; cd = td;
                        int   ti = bi[j]; bi[j] = ci; ci = ti;
                    }
                }
            }
        }
    }
    #pragma unroll
    for (int j = 1; j <= KNN; ++j)
        idxOut[((size_t)b * NTOT + n) * KNN + (j - 1)] = bi[j];
}

// ---------------------------------------------------------------------------
// Fused gather -> [nbr-ctr, ctr] -> GEMM1(f16 WMMA) -> GELU -> GEMM2 -> GELU
// -> mean over K. One wave32 per point-n tile (16 neighbor rows == M tile).
// ---------------------------------------------------------------------------
template<int CIN>
__global__ __launch_bounds__(128) void edge_mlp_kernel(
        const float* __restrict__ fin,      // [B][N][CIN]
        const int*   __restrict__ nbrIdx,   // [B][N][KNN]
        const float* __restrict__ W1,       // [2*CIN][128]
        const float* __restrict__ b1,       // [128]
        const float* __restrict__ W2,       // [128][64]
        const float* __restrict__ b2,       // [64]
        float*       __restrict__ fout,     // [B][N][64]
        const int*   __restrict__ maskOrNull) {
    constexpr int CE   = 2 * CIN;   // edge feature dim (A K-dim of GEMM1)
    constexpr int CH   = 128;       // hidden
    constexpr int COUT = 64;        // output
    constexpr int NW   = 4;         // waves per workgroup
    constexpr int NSPLIT = 8;       // workgroups per batch item
    constexpr int NPG  = NTOT / NSPLIT;

    __shared__ __align__(32) _Float16 sWt1[CH * CE];    // W1^T  [out][in]
    __shared__ __align__(32) _Float16 sWt2[COUT * CH];  // W2^T  [out][in]
    __shared__ float sB1[CH];
    __shared__ float sB2[COUT];
    __shared__ __align__(32) _Float16 sEdge[NW][16 * CE];
    __shared__ __align__(32) _Float16 sH1[NW][16 * CH];

    const int tid  = threadIdx.x;
    const int lane = tid & 31;
    const int w    = tid >> 5;
    const int b    = blockIdx.x / NSPLIT;
    const int nBase = (blockIdx.x % NSPLIT) * NPG;

    // Cooperative transposed weight / bias staging (f16).
    for (int e = tid; e < CE * CH; e += 128) {
        const int i = e / CH, o = e % CH;
        sWt1[o * CE + i] = (_Float16)W1[e];
    }
    for (int e = tid; e < CH * COUT; e += 128) {
        const int i = e / COUT, o = e % COUT;
        sWt2[o * CH + i] = (_Float16)W2[e];
    }
    for (int e = tid; e < CH; e += 128)   sB1[e] = b1[e];
    for (int e = tid; e < COUT; e += 128) sB2[e] = b2[e];
    __syncthreads();

    const int half = lane >> 4;   // 0: lanes 0-15, 1: lanes 16-31
    const int lcol = lane & 15;
    _Float16* edgeW = sEdge[w];
    _Float16* h1W   = sH1[w];

    for (int it = 0; it < NPG / NW; ++it) {
        const int n = nBase + w * (NPG / NW) + it;
        const size_t rowBase = (size_t)b * NTOT + n;

        // ---- build 16 x CE edge tile: [neighbor - center, center] ----
        for (int e = lane; e < 16 * CE; e += 32) {
            const int r = e / CE, c = e % CE;
            float v;
            if (c < CIN) {
                const int nb = nbrIdx[rowBase * KNN + r];
                v = fin[((size_t)b * NTOT + nb) * CIN + c] - fin[rowBase * CIN + c];
            } else {
                v = fin[rowBase * CIN + (c - CIN)];
            }
            edgeW[e] = (_Float16)v;
        }
        asm volatile("s_wait_dscnt 0" ::: "memory");   // intra-wave LDS RAW

        // ---- GEMM1: (16 x CE) @ (CE x 128), K in chunks of 32 ----
        v8f acc1[CH / 16];
        #pragma unroll
        for (int j = 0; j < CH / 16; ++j) acc1[j] = v8f{};

        #pragma unroll
        for (int kc = 0; kc < CE / 32; ++kc) {
            // A fragment (ISA 16-bit 16x32 layout): row = lane&15,
            // lanes<16 hold K [0..7]+[16..23], lanes>=16 hold K [8..15]+[24..31]
            const int kLo = kc * 32 + half * 8;
            v8h alo = *(const v8h*)&edgeW[lcol * CE + kLo];
            v8h ahi = *(const v8h*)&edgeW[lcol * CE + kLo + 16];
            v16h afrag;
            #pragma unroll
            for (int e = 0; e < 8; ++e) { afrag[e] = alo[e]; afrag[e + 8] = ahi[e]; }

            const int kB = kc * 32 + half * 8;   // B: same K order as A
            #pragma unroll
            for (int j = 0; j < CH / 16; ++j) {
                const int col = j * 16 + lcol;
                v8h blo = *(const v8h*)&sWt1[col * CE + kB];
                v8h bhi = *(const v8h*)&sWt1[col * CE + kB + 16];
                v16h bfrag;
                #pragma unroll
                for (int e = 0; e < 8; ++e) { bfrag[e] = blo[e]; bfrag[e + 8] = bhi[e]; }
                acc1[j] = __builtin_amdgcn_wmma_f32_16x16x32_f16(
                    false, afrag, false, bfrag, (short)0, acc1[j], false, false);
            }
        }

        // ---- bias + GELU -> f16 A-staging for GEMM2 ----
        #pragma unroll
        for (int j = 0; j < CH / 16; ++j) {
            const int col  = j * 16 + lcol;
            const float bv = sB1[col];
            #pragma unroll
            for (int r = 0; r < 8; ++r) {
                const int row = r + half * 8;   // C/D layout: lanes>=16 carry M+8
                h1W[row * CH + col] = (_Float16)gelu_exact(acc1[j][r] + bv);
            }
        }
        asm volatile("s_wait_dscnt 0" ::: "memory");

        // ---- GEMM2: (16 x 128) @ (128 x 64) ----
        v8f acc2[COUT / 16];
        #pragma unroll
        for (int j = 0; j < COUT / 16; ++j) acc2[j] = v8f{};

        #pragma unroll
        for (int kc = 0; kc < CH / 32; ++kc) {
            const int kLo = kc * 32 + half * 8;
            v8h alo = *(const v8h*)&h1W[lcol * CH + kLo];
            v8h ahi = *(const v8h*)&h1W[lcol * CH + kLo + 16];
            v16h afrag;
            #pragma unroll
            for (int e = 0; e < 8; ++e) { afrag[e] = alo[e]; afrag[e + 8] = ahi[e]; }

            const int kB = kc * 32 + half * 8;
            #pragma unroll
            for (int j = 0; j < COUT / 16; ++j) {
                const int col = j * 16 + lcol;
                v8h blo = *(const v8h*)&sWt2[col * CH + kB];
                v8h bhi = *(const v8h*)&sWt2[col * CH + kB + 16];
                v16h bfrag;
                #pragma unroll
                for (int e = 0; e < 8; ++e) { bfrag[e] = blo[e]; bfrag[e + 8] = bhi[e]; }
                acc2[j] = __builtin_amdgcn_wmma_f32_16x16x32_f16(
                    false, afrag, false, bfrag, (short)0, acc2[j], false, false);
            }
        }

        // ---- bias + GELU + mean over the 16 neighbor rows ----
        const float maskMul = maskOrNull ? (float)maskOrNull[rowBase] : 1.0f;
        #pragma unroll
        for (int j = 0; j < COUT / 16; ++j) {
            const int col  = j * 16 + lcol;
            const float bv = sB2[col];
            float csum = 0.0f;
            #pragma unroll
            for (int r = 0; r < 8; ++r) csum += gelu_exact(acc2[j][r] + bv);
            csum += __shfl_xor(csum, 16, 32);   // combine row halves across lane 16 split
            if (half == 0)
                fout[rowBase * COUT + col] = csum * (1.0f / 16.0f) * maskMul;
        }
    }
}

// ---------------------------------------------------------------------------
extern "C" void kernel_launch(void* const* d_in, const int* in_sizes, int n_in,
                              void* d_out, int out_size, void* d_ws, size_t ws_size,
                              hipStream_t stream) {
    const float* x      = (const float*)d_in[0];
    const float* points = (const float*)d_in[1];
    const int*   mask   = (const int*)d_in[2];
    const float* W1_0   = (const float*)d_in[3];
    const float* b1_0   = (const float*)d_in[4];
    const float* W2_0   = (const float*)d_in[5];
    const float* b2_0   = (const float*)d_in[6];
    const float* W1_1   = (const float*)d_in[7];
    const float* b1_1   = (const float*)d_in[8];
    const float* W2_1   = (const float*)d_in[9];
    const float* b2_1   = (const float*)d_in[10];

    float* feats1 = (float*)d_ws;                                     // B*N*64 f32
    int*   idxBuf = (int*)((char*)d_ws +
                           (size_t)BATCH * NTOT * 64 * sizeof(float)); // B*N*16 i32

    const dim3 knnGrid(BATCH * (NTOT / 256));
    const dim3 mlpGrid(BATCH * 8);

    // Block 0: knn on shifted points (F=3), MLP 64->128->64 on x.
    knn_kernel<3><<<knnGrid, 256, 0, stream>>>(points, mask, idxBuf);
    edge_mlp_kernel<32><<<mlpGrid, 128, 0, stream>>>(x, idxBuf, W1_0, b1_0, W2_0, b2_0,
                                                     feats1, nullptr);
    // Block 1: knn in 64-d feature space, MLP 128->128->64, final mask multiply.
    knn_kernel<64><<<knnGrid, 256, 0, stream>>>(feats1, mask, idxBuf);
    edge_mlp_kernel<64><<<mlpGrid, 128, 0, stream>>>(feats1, idxBuf, W1_1, b1_1, W2_1, b2_1,
                                                     (float*)d_out, mask);
}
